// PhysicsAttention_5952824673231
// MI455X (gfx1250) — matrix operand order repaired and checked by reading, not compiled, not run on hardware
//
#include <hip/hip_runtime.h>
#include <hip/hip_bf16.h>

typedef _Float16 f16;
typedef __attribute__((ext_vector_type(16))) _Float16 v16h;
typedef __attribute__((ext_vector_type(8)))  _Float16 v8h;
typedef __attribute__((ext_vector_type(8)))  float    v8f;

union FragAB { v16h v; v8h h8[2]; _Float16 h[16]; };
union FragC  { v8f  v; float f[8]; };

// ---------------------------------------------------------------------------
// WMMA helpers (wave32, 16x16x32 f16 -> f32). Layouts per CDNA5 ISA 7.12.2:
//   A (16x32): lane holds row m=lane&15; halfs 0..7 -> K = 8*(lane>>4)+i,
//              halfs 8..15 -> K = 16 + 8*(lane>>4) + (i-8)
//   B (32x16): lane holds col n=lane&15; same K mapping as A
//   C (16x16): lane col n=lane&15; vgpr j -> row m = j + 8*(lane>>4)
// ---------------------------------------------------------------------------
__device__ __forceinline__ v16h load_frag_a(const f16* base, int pitch, int lane) {
  FragAB a;
  int m  = lane & 15;
  int kb = (lane >> 4) << 3;            // 0 or 8
  const f16* p = base + m * pitch;
  a.h8[0] = *(const v8h*)(p + kb);      // K = kb .. kb+7
  a.h8[1] = *(const v8h*)(p + 16 + kb); // K = 16+kb .. 16+kb+7
  return a.v;
}

__device__ __forceinline__ v16h load_frag_b(const f16* base, int pitch, int lane) {
  // base points at [n][k] storage (i.e. transposed weight / K-major per column)
  FragAB b;
  int n  = lane & 15;
  int kb = (lane >> 4) << 3;            // 0 or 8
  const f16* p = base + n * pitch + kb;
  b.h8[0] = *(const v8h*)(p);
  b.h8[1] = *(const v8h*)(p + 16);
  return b.v;
}

__device__ __forceinline__ v8f wmma32(v16h a, v16h b, v8f c) {
  return __builtin_amdgcn_wmma_f32_16x16x32_f16(false, a, false, b, (short)0, c,
                                                false, false);
}

// ---------------------------------------------------------------------------
// CDNA5 async global->LDS copy (ASYNCcnt path).  One instruction per wave
// moves 16 B per lane straight into LDS (no VGPR round trip).  LDS byte
// address = low 32 bits of the flat shared-pointer (ISA 10.2: LDS aperture
// keeps the LDS offset in addr[31:0]).  Inline asm is used because the TDM /
// async builtins differ in arity across ROCm 7.2 and upstream toolchains.
// ---------------------------------------------------------------------------
__device__ __forceinline__ void async_b128(f16* lds_dst, const f16* gsrc) {
  unsigned l = (unsigned)(unsigned long long)(size_t)lds_dst;
  asm volatile("global_load_async_to_lds_b128 %0, %1, off"
               :: "v"(l), "v"(gsrc) : "memory");
}
__device__ __forceinline__ void wait_async_le3() {
  asm volatile("s_wait_asynccnt 0x3" ::: "memory");
}
__device__ __forceinline__ void wait_async_0() {
  asm volatile("s_wait_asynccnt 0x0" ::: "memory");
}

// ---------------------------------------------------------------------------
// Prep kernels
// ---------------------------------------------------------------------------
__global__ void k_f32_to_f16(const float* __restrict__ src, f16* __restrict__ dst, int n) {
  int i = blockIdx.x * blockDim.x + threadIdx.x;
  if (i < n) dst[i] = (f16)src[i];
}

// dst[n*K + k] = src[k*N + n]   (store W^T so it is a ready-made B operand)
__global__ void k_transpose_f16(const float* __restrict__ src, f16* __restrict__ dst,
                                int K, int N) {
  int i = blockIdx.x * blockDim.x + threadIdx.x;
  if (i < K * N) {
    int k = i / N, n = i - k * N;
    dst[n * K + k] = (f16)src[i];
  }
}

__global__ void k_zero_f32(float* __restrict__ p, int n) {
  int i = blockIdx.x * blockDim.x + threadIdx.x;
  if (i < n) p[i] = 0.0f;
}

// ---------------------------------------------------------------------------
// Generic WMMA GEMM + bias: C[M][N] = A[M][K] * W[K][N] + bias,  W given as
// Bt[N][K].  Block tile 128(M) x 64(N), K-chunk 32.  256 threads = 8 waves,
// wave w owns rows [w*16, w*16+16) and all four 16-wide N subtiles.
// Double-buffered LDS filled by async global->LDS copies: while WMMA consumes
// buffer `buf`, the next K-chunk streams into `buf^1` (3 async instructions
// per wave per chunk -> s_wait_asynccnt 3 drains exactly the current chunk,
// since async loads complete in order).
// ---------------------------------------------------------------------------
template <typename OutT>
__global__ void k_gemm_bias(const f16* __restrict__ A, const f16* __restrict__ Bt,
                            const float* __restrict__ bias, OutT* __restrict__ C,
                            int M, int N, int K) {
  __shared__ f16 As[2][128 * 40];   // pitch 40 halfs (16B-aligned, pad vs conflicts)
  __shared__ f16 Bs[2][64 * 40];
  int t = threadIdx.x, lane = t & 31, w = t >> 5;
  long m0 = (long)blockIdx.x * 128;
  int  n0 = blockIdx.y * 64;
  FragC acc[4] = {};

  auto fill = [&](int buf, int k0) {
    // A tile: 128 rows x 32 halfs = 512 x 16B, 2 per thread
    #pragma unroll
    for (int i = 0; i < 2; i++) {
      int l = t + i * 256;
      int row = l >> 2, ko = (l & 3) * 8;
      async_b128(&As[buf][row * 40 + ko], A + (m0 + row) * K + k0 + ko);
    }
    // B tile: 64 rows(n) x 32 halfs(k) = 256 x 16B, 1 per thread
    int row = t >> 2, ko = (t & 3) * 8;
    async_b128(&Bs[buf][row * 40 + ko], Bt + (long)(n0 + row) * K + k0 + ko);
  };

  int nk = K >> 5;
  fill(0, 0);
  for (int ki = 0; ki < nk; ki++) {
    int buf = ki & 1;
    __syncthreads();                       // everyone done reading As[buf^1]
    if (ki + 1 < nk) {
      fill(buf ^ 1, (ki + 1) << 5);        // prefetch next chunk
      wait_async_le3();                    // drain current chunk only
    } else {
      wait_async_0();
    }
    __syncthreads();                       // current chunk visible to all waves
    v16h a = load_frag_a(&As[buf][(w * 16) * 40], 40, lane);
    #pragma unroll
    for (int s = 0; s < 4; s++) {
      v16h b = load_frag_b(&Bs[buf][(s * 16) * 40], 40, lane);
      acc[s].v = wmma32(a, b, acc[s].v);
    }
  }
  int col = lane & 15, half = lane >> 4;
  #pragma unroll
  for (int s = 0; s < 4; s++) {
    int cg = n0 + s * 16 + col;
    float bv = bias[cg];
    #pragma unroll
    for (int j = 0; j < 8; j++) {
      long row = m0 + w * 16 + j + half * 8;
      C[row * N + cg] = (OutT)(acc[s].f[j] + bv);
    }
  }
}

// ---------------------------------------------------------------------------
// Slice logits + softmax over G (fused, per head).
// xm: [b][n][h*64+d] fp16.  WsT: [g][d] fp16 (W_slice^T).  Output sw:
// [(b*H+h)*N+n][g] fp16, plus atomic partial sums for slice_norm.
// Block = 128 rows of one (b,h); tiles staged by async global->LDS copies.
// Row softmax in-register with shfl_xor across the 16-lane half that owns
// each row (wave32 C-layout), 1/x via v_rcp_f32.
// ---------------------------------------------------------------------------
__global__ void k_slice_softmax(const f16* __restrict__ xm, const f16* __restrict__ WsT,
                                const float* __restrict__ b_slice,
                                const float* __restrict__ temperature,
                                f16* __restrict__ sw, float* __restrict__ snorm, int N) {
  __shared__ f16 As[128 * 72];
  __shared__ f16 Bs[64 * 72];
  int t = threadIdx.x, lane = t & 31, w = t >> 5;
  int bh = blockIdx.y, b = bh >> 3, h = bh & 7;
  long nb = (long)blockIdx.x * 128;
  // stage x_mid head tile: 128 x 64 halfs = 1024 x 16B (4/thread), async
  #pragma unroll
  for (int i = 0; i < 4; i++) {
    int l = t + i * 256;
    int row = l >> 3, ko = (l & 7) * 8;
    async_b128(&As[row * 72 + ko],
               xm + ((long)b * N + nb + row) * 512 + h * 64 + ko);
  }
  // stage W_slice^T: 64 x 64 = 512 x 16B (2/thread), async
  #pragma unroll
  for (int i = 0; i < 2; i++) {
    int l = t + i * 256;
    int row = l >> 3, ko = (l & 7) * 8;
    async_b128(&Bs[row * 72 + ko], WsT + row * 64 + ko);
  }
  wait_async_0();
  __syncthreads();
  FragC acc[4] = {};
  #pragma unroll
  for (int kc = 0; kc < 64; kc += 32) {
    v16h a = load_frag_a(&As[(w * 16) * 72 + kc], 72, lane);
    #pragma unroll
    for (int s = 0; s < 4; s++) {
      v16h bb = load_frag_b(&Bs[(s * 16) * 72 + kc], 72, lane);
      acc[s].v = wmma32(a, bb, acc[s].v);
    }
  }
  float tv = temperature[h];
  tv = fminf(fmaxf(tv, 0.1f), 5.0f);
  float invt = __builtin_amdgcn_rcpf(tv);
  int col = lane & 15, half = lane >> 4;
  float bsl[4];
  #pragma unroll
  for (int s = 0; s < 4; s++) bsl[s] = b_slice[s * 16 + col];
  float csum[4] = {0.f, 0.f, 0.f, 0.f};
  #pragma unroll
  for (int j = 0; j < 8; j++) {
    float lg[4];
    float mx = -1e30f;
    #pragma unroll
    for (int s = 0; s < 4; s++) {
      lg[s] = (acc[s].f[j] + bsl[s]) * invt;
      mx = fmaxf(mx, lg[s]);
    }
    for (int msk = 1; msk < 16; msk <<= 1) mx = fmaxf(mx, __shfl_xor(mx, msk, 32));
    float ss = 0.f;
    #pragma unroll
    for (int s = 0; s < 4; s++) { lg[s] = __expf(lg[s] - mx); ss += lg[s]; }
    for (int msk = 1; msk < 16; msk <<= 1) ss += __shfl_xor(ss, msk, 32);
    float inv = __builtin_amdgcn_rcpf(ss);
    long row = (long)bh * N + nb + w * 16 + j + half * 8;
    #pragma unroll
    for (int s = 0; s < 4; s++) {
      float wv = lg[s] * inv;
      csum[s] += wv;
      sw[row * 64 + s * 16 + col] = (f16)wv;
    }
  }
  // per-wave partial column sums -> slice_norm (pair lanes L, L^16 share a col)
  #pragma unroll
  for (int s = 0; s < 4; s++) {
    float v2 = csum[s] + __shfl_xor(csum[s], 16, 32);
    if (lane < 16) atomicAdd(&snorm[bh * 64 + s * 16 + col], v2);
  }
}

// ---------------------------------------------------------------------------
// slice_token[g][c] = sum_n sw[n][g] * fx[n][c]   (split-K over N, atomics)
// Both operands staged transposed into LDS so fragments load contiguously.
// ---------------------------------------------------------------------------
__global__ void k_slice_token(const f16* __restrict__ sw, const f16* __restrict__ fx,
                              float* __restrict__ stoken, int N, int NC) {
  __shared__ f16 swT[64 * 40];  // [g][n]
  __shared__ f16 fxT[64 * 40];  // [c][n]
  int t = threadIdx.x, lane = t & 31, w = t >> 5;
  int bh = blockIdx.y, b = bh >> 3, h = bh & 7;
  long nbase = (long)blockIdx.x * NC;
  int mt = w >> 1, n0a = (w & 1) * 32;
  FragC acc[2] = {};
  for (long nb = nbase; nb < nbase + NC; nb += 32) {
    __syncthreads();
    for (int e = t; e < 2048; e += 256) {
      int n = e >> 6, g = e & 63;
      swT[g * 40 + n] = sw[((long)bh * N + nb + n) * 64 + g];
      fxT[g * 40 + n] = fx[((long)b * N + nb + n) * 512 + h * 64 + g];
    }
    __syncthreads();
    v16h a = load_frag_a(&swT[(mt * 16) * 40], 40, lane);
    #pragma unroll
    for (int s = 0; s < 2; s++) {
      v16h bb = load_frag_b(&fxT[(n0a + s * 16) * 40], 40, lane);
      acc[s].v = wmma32(a, bb, acc[s].v);
    }
  }
  int col = lane & 15, half = lane >> 4;
  #pragma unroll
  for (int s = 0; s < 2; s++)
    #pragma unroll
    for (int j = 0; j < 8; j++) {
      int g = mt * 16 + j + half * 8;
      int c = n0a + s * 16 + col;
      atomicAdd(&stoken[(long)bh * 4096 + g * 64 + c], acc[s].f[j]);
    }
}

// ---------------------------------------------------------------------------
// Per-(b,h) 64x64 attention among slice tokens: normalize, q/k/v, softmax, o.
// ---------------------------------------------------------------------------
#define SCALE_QK 0.125f
__global__ void k_attn(const float* __restrict__ stoken, const float* __restrict__ snorm,
                       const f16* __restrict__ WqT, const f16* __restrict__ WkT,
                       const f16* __restrict__ WvT, f16* __restrict__ oslice) {
  __shared__ f16 st[64 * 72];   // [g][c]
  __shared__ f16 qs[64 * 72];   // [g][c]
  __shared__ f16 ks[64 * 72];   // [g'][c]  (already B-operand layout for q.k^T)
  __shared__ f16 vT[64 * 72];   // [c][g']  (B-operand layout for attn.v)
  __shared__ f16 at[64 * 72];   // [g][g']
  int bh = blockIdx.x;
  int t = threadIdx.x, lane = t & 31, w = t >> 5;
  int col = lane & 15, half = lane >> 4;
  // 1) normalize pooled tokens
  for (int e = t; e < 4096; e += 256) {
    int g = e >> 6, c = e & 63;
    float v = stoken[(long)bh * 4096 + e] *
              __builtin_amdgcn_rcpf(snorm[bh * 64 + g] + 1e-5f);
    st[g * 72 + c] = (f16)v;
  }
  __syncthreads();
  // 2) q,k,v  (wave -> 2 tiles each; weight B-fragments straight from global)
  int mt = w >> 1, n0a = (w & 1) * 32;
  {
    FragC aq[2] = {}, ak2[2] = {}, av[2] = {};
    #pragma unroll
    for (int kc = 0; kc < 64; kc += 32) {
      v16h a = load_frag_a(&st[(mt * 16) * 72 + kc], 72, lane);
      #pragma unroll
      for (int s = 0; s < 2; s++) {
        int n0 = n0a + s * 16;
        aq[s].v  = wmma32(a, load_frag_b(WqT + n0 * 64 + kc, 64, lane), aq[s].v);
        ak2[s].v = wmma32(a, load_frag_b(WkT + n0 * 64 + kc, 64, lane), ak2[s].v);
        av[s].v  = wmma32(a, load_frag_b(WvT + n0 * 64 + kc, 64, lane), av[s].v);
      }
    }
    #pragma unroll
    for (int s = 0; s < 2; s++)
      #pragma unroll
      for (int j = 0; j < 8; j++) {
        int row = mt * 16 + j + half * 8;
        int cc  = n0a + s * 16 + col;
        qs[row * 72 + cc] = (f16)aq[s].f[j];
        ks[row * 72 + cc] = (f16)ak2[s].f[j];
        vT[cc * 72 + row] = (f16)av[s].f[j];   // store v transposed
      }
  }
  __syncthreads();
  // 3) attn = softmax(q k^T * scale): waves 0..3 own one 16-row tile each so a
  //    full 64-wide row lives in one wave -> shfl softmax.
  if (w < 4) {
    FragC lg[4] = {};
    #pragma unroll
    for (int kc = 0; kc < 64; kc += 32) {
      v16h a = load_frag_a(&qs[(w * 16) * 72 + kc], 72, lane);
      #pragma unroll
      for (int s = 0; s < 4; s++)
        lg[s].v = wmma32(a, load_frag_b(&ks[(s * 16) * 72 + kc], 72, lane), lg[s].v);
    }
    #pragma unroll
    for (int j = 0; j < 8; j++) {
      float e[4];
      float mx = -1e30f;
      #pragma unroll
      for (int s = 0; s < 4; s++) { e[s] = lg[s].f[j] * SCALE_QK; mx = fmaxf(mx, e[s]); }
      for (int msk = 1; msk < 16; msk <<= 1) mx = fmaxf(mx, __shfl_xor(mx, msk, 32));
      float ss = 0.f;
      #pragma unroll
      for (int s = 0; s < 4; s++) { e[s] = __expf(e[s] - mx); ss += e[s]; }
      for (int msk = 1; msk < 16; msk <<= 1) ss += __shfl_xor(ss, msk, 32);
      float inv = __builtin_amdgcn_rcpf(ss);
      int row = w * 16 + j + half * 8;
      #pragma unroll
      for (int s = 0; s < 4; s++) at[row * 72 + s * 16 + col] = (f16)(e[s] * inv);
    }
  }
  __syncthreads();
  // 4) out_slice = attn @ v
  {
    FragC o[2] = {};
    #pragma unroll
    for (int kc = 0; kc < 64; kc += 32) {
      v16h a = load_frag_a(&at[(mt * 16) * 72 + kc], 72, lane);
      #pragma unroll
      for (int s = 0; s < 2; s++)
        o[s].v = wmma32(a, load_frag_b(&vT[(n0a + s * 16) * 72 + kc], 72, lane), o[s].v);
    }
    #pragma unroll
    for (int s = 0; s < 2; s++)
      #pragma unroll
      for (int j = 0; j < 8; j++) {
        int row = mt * 16 + j + half * 8;
        oslice[(long)bh * 4096 + row * 64 + n0a + s * 16 + col] = (f16)o[s].f[j];
      }
  }
}

// ---------------------------------------------------------------------------
// out_x[n][h*64+c] = sum_g sw[n][g] * out_slice[g][c]
// A fragments loaded straight from global (row-major, pitch 64).
// ---------------------------------------------------------------------------
__global__ void k_scatter(const f16* __restrict__ sw, const f16* __restrict__ oslice,
                          f16* __restrict__ outx, int N) {
  __shared__ f16 oT[64 * 72];  // [c][g]
  int bh = blockIdx.y, b = bh >> 3, h = bh & 7;
  long nb = (long)blockIdx.x * 64;
  int t = threadIdx.x, lane = t & 31, w = t >> 5;
  for (int e = t; e < 4096; e += 256) {
    int g = e >> 6, c = e & 63;
    oT[c * 72 + g] = oslice[(long)bh * 4096 + e];
  }
  __syncthreads();
  int mt = w >> 1, n0a = (w & 1) * 32, col = lane & 15, half = lane >> 4;
  FragC acc[2] = {};
  const f16* abase = sw + ((long)bh * N + nb + mt * 16) * 64;
  #pragma unroll
  for (int kc = 0; kc < 64; kc += 32) {
    v16h a = load_frag_a(abase + kc, 64, lane);
    #pragma unroll
    for (int s = 0; s < 2; s++)
      acc[s].v = wmma32(a, load_frag_b(&oT[(n0a + s * 16) * 72 + kc], 72, lane), acc[s].v);
  }
  #pragma unroll
  for (int s = 0; s < 2; s++)
    #pragma unroll
    for (int j = 0; j < 8; j++) {
      long row = (long)b * N + nb + mt * 16 + j + half * 8;
      outx[row * 512 + h * 64 + n0a + s * 16 + col] = (f16)acc[s].f[j];
    }
}

// ---------------------------------------------------------------------------
extern "C" void kernel_launch(void* const* d_in, const int* in_sizes, int n_in,
                              void* d_out, int out_size, void* d_ws, size_t ws_size,
                              hipStream_t stream) {
  const int B = 4, N = 16384, DIM = 256, H = 8, INNER = 512;
  const long M = (long)B * N;                   // 65536
  const float* x       = (const float*)d_in[0];
  const float* W_fx    = (const float*)d_in[1];
  const float* b_fx    = (const float*)d_in[2];
  const float* W_x     = (const float*)d_in[3];
  const float* b_x     = (const float*)d_in[4];
  const float* W_slice = (const float*)d_in[5];
  const float* b_slice = (const float*)d_in[6];
  const float* W_q     = (const float*)d_in[7];
  const float* W_k     = (const float*)d_in[8];
  const float* W_v     = (const float*)d_in[9];
  const float* W_out   = (const float*)d_in[10];
  const float* b_out   = (const float*)d_in[11];
  const float* temp    = (const float*)d_in[12];

  char* p = (char*)d_ws;
  auto alloc = [&](size_t bytes) -> void* {
    void* r = (void*)p;
    p += (bytes + 255) & ~(size_t)255;
    return r;
  };
  f16* x16   = (f16*)alloc(M * DIM * sizeof(f16));      // 32 MB
  f16* fx16  = (f16*)alloc(M * INNER * sizeof(f16));    // 64 MB
  f16* xm16  = (f16*)alloc(M * INNER * sizeof(f16));    // 64 MB (reused as out_x)
  f16* sw16  = (f16*)alloc((size_t)B * H * N * 64 * sizeof(f16)); // 64 MB
  f16* WfxT  = (f16*)alloc(512 * 256 * sizeof(f16));
  f16* WxT   = (f16*)alloc(512 * 256 * sizeof(f16));
  f16* WsT   = (f16*)alloc(64 * 64 * sizeof(f16));
  f16* WqT   = (f16*)alloc(64 * 64 * sizeof(f16));
  f16* WkT   = (f16*)alloc(64 * 64 * sizeof(f16));
  f16* WvT   = (f16*)alloc(64 * 64 * sizeof(f16));
  f16* WoutT = (f16*)alloc(256 * 512 * sizeof(f16));
  float* snorm  = (float*)alloc(32 * 64 * sizeof(float));
  float* stoken = (float*)alloc(32 * 4096 * sizeof(float));
  f16* osl16 = (f16*)alloc(32 * 4096 * sizeof(f16));
  f16* outx16 = xm16;  // x_mid is dead after k_slice_softmax

  // --- prep ---
  int nx = (int)(M * DIM);
  k_f32_to_f16<<<(nx + 255) / 256, 256, 0, stream>>>(x, x16, nx);
  k_transpose_f16<<<(256 * 512 + 255) / 256, 256, 0, stream>>>(W_fx, WfxT, 256, 512);
  k_transpose_f16<<<(256 * 512 + 255) / 256, 256, 0, stream>>>(W_x, WxT, 256, 512);
  k_transpose_f16<<<(4096 + 255) / 256, 256, 0, stream>>>(W_slice, WsT, 64, 64);
  k_transpose_f16<<<(4096 + 255) / 256, 256, 0, stream>>>(W_q, WqT, 64, 64);
  k_transpose_f16<<<(4096 + 255) / 256, 256, 0, stream>>>(W_k, WkT, 64, 64);
  k_transpose_f16<<<(4096 + 255) / 256, 256, 0, stream>>>(W_v, WvT, 64, 64);
  k_transpose_f16<<<(512 * 256 + 255) / 256, 256, 0, stream>>>(W_out, WoutT, 512, 256);
  k_zero_f32<<<(2048 + 255) / 256, 256, 0, stream>>>(snorm, 2048);
  k_zero_f32<<<(131072 + 255) / 256, 256, 0, stream>>>(stoken, 131072);

  // --- pipeline ---
  k_gemm_bias<f16><<<dim3(512, 8), 256, 0, stream>>>(x16, WfxT, b_fx, fx16,
                                                     (int)M, 512, 256);
  k_gemm_bias<f16><<<dim3(512, 8), 256, 0, stream>>>(x16, WxT, b_x, xm16,
                                                     (int)M, 512, 256);
  k_slice_softmax<<<dim3(N / 128, 32), 256, 0, stream>>>(xm16, WsT, b_slice, temp,
                                                         sw16, snorm, N);
  k_slice_token<<<dim3(16, 32), 256, 0, stream>>>(sw16, fx16, stoken, N, N / 16);
  k_attn<<<32, 256, 0, stream>>>(stoken, snorm, WqT, WkT, WvT, osl16);
  k_scatter<<<dim3(N / 64, 32), 256, 0, stream>>>(sw16, osl16, outx16, N);
  k_gemm_bias<float><<<dim3(512, 4), 256, 0, stream>>>(outx16, WoutT, b_out,
                                                       (float*)d_out, (int)M, 256, 512);
}
